// ContributionRNN_79293686219377
// MI455X (gfx1250) — matrix operand order repaired and checked by reading, not compiled, not run on hardware
//
#include <hip/hip_runtime.h>

// ContributionRNN for MI455X (gfx1250, wave32, WMMA + TDM).
// Pipeline:
//   1) init:    zero sync counter + out vector, fuse b_ih2+b_hh2
//   2) prep_w:  W_ih1 -> Wx(bf16), Wh(bf16); W_ih2 -> W2(bf16)
//   3) prep_x:  x -> xbf (bf16)            (so the GEMM loop is load+WMMA only)
//   4) gemm:    pre_x = x @ Wx^T + (b_ih1+b_hh1)  (bf16 WMMA, 16x64 per wave)
//   5) seq:     persistent 4-block kernel; weights staged into LDS by the
//               Tensor Data Mover (padded rows via D# pad fields); per step
//               two LDS-resident WMMA matvecs + tanh; cross-WGP exchange via
//               L2 + atomic release/acquire barrier; final fc on block 0.
// Workspace requirement: ~50 MB.

#define SEQ   16384
#define HID   512
#define NBLK  4
#define RPB   128            // output rows owned per block
#define LROW  520            // LDS row stride (bf16 elems): 512 + 8 pad

typedef float          v8f   __attribute__((ext_vector_type(8)));
typedef __bf16         v8bf  __attribute__((ext_vector_type(8)));
typedef __bf16         v16bf __attribute__((ext_vector_type(16)));
typedef unsigned short v8us  __attribute__((ext_vector_type(8)));
typedef unsigned int   u32x4 __attribute__((ext_vector_type(4)));
typedef int            i32x8 __attribute__((ext_vector_type(8)));
typedef int            i32x4 __attribute__((ext_vector_type(4)));

__device__ inline unsigned short f2us_bf(float f) {       // f32 -> bf16 (RNE)
  unsigned u = __float_as_uint(f);
  u += 0x7fffu + ((u >> 16) & 1u);
  return (unsigned short)(u >> 16);
}
__device__ inline float us2f_bf(unsigned short s) {
  return __uint_as_float(((unsigned)s) << 16);
}
__device__ inline v16bf cat16(v8us lo, v8us hi) {
  v8bf a = __builtin_bit_cast(v8bf, lo);
  v8bf b = __builtin_bit_cast(v8bf, hi);
  return __builtin_shufflevector(a, b, 0,1,2,3,4,5,6,7,8,9,10,11,12,13,14,15);
}

// ------------------------------------------------------------- TDM load ----
// 2D tile (512 x rows) of 2-byte elems, global->LDS, with LDS padding of
// 4 dwords after every 256 dwords => effective LDS row stride = 520 bf16.
__device__ inline void tdm_load_weights(unsigned lds_byte_off,
                                        const unsigned short* gsrc,
                                        unsigned rows) {
  unsigned long long ga = (unsigned long long)(const void*)gsrc;
  u32x4 g0;
  g0[0] = 1u;                                        // count=1, user mode
  g0[1] = lds_byte_off;                              // D#.lds_addr
  g0[2] = (unsigned)(ga & 0xffffffffu);              // D#.global_addr lo
  g0[3] = (unsigned)((ga >> 32) & 0x01ffffffu)       // D#.global_addr hi
        | (2u << 30);                                // D#.type = 2 (image)
  i32x8 g1;
  g1[0] = (int)((1u << 16)                           // data_size = 2 bytes
              | (1u << 20)                           // pad_enable
              | (7u << 22)                           // pad_interval = 256 dw
              | (3u << 25));                         // pad_amount   = 4 dw
  g1[1] = (int)((512u & 0xffffu) << 16);             // tensor_dim0 lo16
  g1[2] = (int)(512u << 16);                         // dim0 hi16 | dim1 lo16
  g1[3] = (int)(512u << 16);                         // dim1 hi16 | tile_dim0
  g1[4] = (int)(rows & 0xffffu);                     // tile_dim1 | tile_dim2=0
  g1[5] = (int)512;                                  // tensor_dim0_stride lo32
  g1[6] = 0;                                         // stride0 hi | stride1 lo
  g1[7] = 0;
  i32x4 z4 = {0, 0, 0, 0};
  i32x8 z8 = {0, 0, 0, 0, 0, 0, 0, 0};
  // 6-arg form (clang-23 / therock-10.0 headers): groups 2,3 unused (<=2D).
  __builtin_amdgcn_tensor_load_to_lds(g0, g1, z4, z4, z8, 0);
}

// ---------------------------------------------------------------- init ----
__global__ void init_kernel(unsigned* counter, unsigned* out_u32,
                            float* c2, const float* b_ih2, const float* b_hh2) {
  const int t = threadIdx.x;
  if (t == 0) *counter = 0u;
  if (t < HID / 2) out_u32[t] = 0u;        // out0 = zeros (bf16)
  if (t < HID) c2[t] = b_ih2[t] + b_hh2[t];
}

// ---------------------------------------------------------------- prep ----
__global__ __launch_bounds__(256) void prep_w_kernel(
    const float* __restrict__ W_ih1, const float* __restrict__ W_ih2,
    unsigned short* __restrict__ Wx, unsigned short* __restrict__ Wh,
    unsigned short* __restrict__ W2) {
  const int idx = blockIdx.x * 256 + threadIdx.x;   // 0 .. 512*512-1
  const int n = idx >> 9, k = idx & 511;
  Wx[idx] = f2us_bf(W_ih1[n * 1024 + k]);
  Wh[idx] = f2us_bf(W_ih1[n * 1024 + 512 + k]);
  W2[idx] = f2us_bf(W_ih2[idx]);
}

__global__ __launch_bounds__(256) void prep_x_kernel(
    const float* __restrict__ x, unsigned short* __restrict__ xbf) {
  const int idx = blockIdx.x * 256 + threadIdx.x;   // 0 .. SEQ*HID-1
  xbf[idx] = f2us_bf(x[idx]);
}

// ------------------------------------------------- pre_x = x@Wx^T + c1 ----
// One wave computes a 16(M) x 64(N) strip: A fragment reused by 4 WMMAs.
__global__ __launch_bounds__(256) void gemm_prex(
    const unsigned short* __restrict__ xbf,
    const unsigned short* __restrict__ Wx,
    const float* __restrict__ b_ih1, const float* __restrict__ b_hh1,
    float* __restrict__ pre) {
  const int lane = threadIdx.x & 31;
  const int wave = threadIdx.x >> 5;
  const int tm = blockIdx.x;            // 0..1023  (16-row strip)
  const int ng = wave;                  // 0..7     (64-col group)
  const int q  = lane & 15;
  const int hi = lane >> 4;
  const unsigned short* xr = xbf + (size_t)(tm * 16 + q) * HID;   // A row M=q
  const unsigned short* wr = Wx  + (size_t)(ng * 64 + q) * HID;   // B col N=q
  v8f acc0 = {0.f,0.f,0.f,0.f,0.f,0.f,0.f,0.f};
  v8f acc1 = acc0, acc2 = acc0, acc3 = acc0;
#pragma unroll
  for (int kb = 0; kb < HID / 32; ++kb) {
    // A frag: lane holds 16 contiguous K of its row (lanes 16-31 +16)
    const int ka = kb * 32 + hi * 16;
    v16bf a = cat16(*(const v8us*)(xr + ka), *(const v8us*)(xr + ka + 8));
    // B frags: lane holds 16 contiguous K of its column
    const int kn = kb * 32 + hi * 16;
    v16bf b0 = cat16(*(const v8us*)(wr + 0 * 16 * HID + kn),
                     *(const v8us*)(wr + 0 * 16 * HID + kn + 8));
    v16bf b1 = cat16(*(const v8us*)(wr + 1 * 16 * HID + kn),
                     *(const v8us*)(wr + 1 * 16 * HID + kn + 8));
    v16bf b2 = cat16(*(const v8us*)(wr + 2 * 16 * HID + kn),
                     *(const v8us*)(wr + 2 * 16 * HID + kn + 8));
    v16bf b3 = cat16(*(const v8us*)(wr + 3 * 16 * HID + kn),
                     *(const v8us*)(wr + 3 * 16 * HID + kn + 8));
    acc0 = __builtin_amdgcn_wmma_f32_16x16x32_bf16(false, a, false, b0,
                                                   (short)0, acc0, false, false);
    acc1 = __builtin_amdgcn_wmma_f32_16x16x32_bf16(false, a, false, b1,
                                                   (short)0, acc1, false, false);
    acc2 = __builtin_amdgcn_wmma_f32_16x16x32_bf16(false, a, false, b2,
                                                   (short)0, acc2, false, false);
    acc3 = __builtin_amdgcn_wmma_f32_16x16x32_bf16(false, a, false, b3,
                                                   (short)0, acc3, false, false);
  }
  v8f accs[4] = {acc0, acc1, acc2, acc3};
#pragma unroll
  for (int j = 0; j < 4; ++j) {
    const int n = ng * 64 + j * 16 + q;
    const float bias = b_ih1[n] + b_hh1[n];
#pragma unroll
    for (int r = 0; r < 8; ++r) {               // D: M = r + 8*hi, N = q
      const int m = tm * 16 + hi * 8 + r;
      pre[(size_t)m * HID + n] = accs[j][r] + bias;
    }
  }
}

// ----------------------------------------------------- sequential part ----
#define SMEM_SEQ (RPB * LROW * 2 * 2 + HID * 2 + RPB * 4 + 256 * 4)  // 268800 B

__device__ inline void grid_barrier(unsigned* counter, unsigned target) {
  __syncthreads();
  if (threadIdx.x == 0) {
    __hip_atomic_fetch_add(counter, 1u, __ATOMIC_RELEASE, __HIP_MEMORY_SCOPE_AGENT);
    while (__hip_atomic_load(counter, __ATOMIC_ACQUIRE, __HIP_MEMORY_SCOPE_AGENT) < target)
      __builtin_amdgcn_s_sleep(1);
  }
  __syncthreads();
}

__global__ __launch_bounds__(256) void rnn_seq(
    const float* __restrict__ pre,
    const unsigned short* __restrict__ Whg,
    const unsigned short* __restrict__ W2g,
    const float* __restrict__ c2g,
    unsigned short* outb, unsigned short* h1b, unsigned* counter,
    const float* __restrict__ Wfc, const float* __restrict__ bfc,
    float* __restrict__ out) {
  extern __shared__ char smem[];
  unsigned short* Wh_l = (unsigned short*)smem;          // RPB x LROW bf16
  unsigned short* W2_l = Wh_l + RPB * LROW;              // RPB x LROW bf16
  unsigned short* v_l  = W2_l + RPB * LROW;              // HID bf16 (vector)
  float* c2_l = (float*)(v_l + HID);                     // RPB f32
  float* red  = c2_l + RPB;                              // 256 f32
  const int tid = threadIdx.x;
  const int lane = tid & 31, wave = tid >> 5;
  const int g = blockIdx.x;

  // Stage this block's weight slices into LDS via the Tensor Data Mover;
  // D# pad fields insert 16B per 1024B row -> 520-element padded rows.
  if (wave == 0) {
    tdm_load_weights(0u,              Whg + (size_t)g * RPB * HID, RPB);
    tdm_load_weights(RPB * LROW * 2u, W2g + (size_t)g * RPB * HID, RPB);
    __builtin_amdgcn_s_wait_tensorcnt(0);
  }
  if (tid < RPB) c2_l[tid] = c2g[g * RPB + tid];
  __syncthreads();

  const int q = lane & 15, hi = lane >> 4;
  const int rowl = wave * 16 + hi * 8;                   // first of 8 D rows
  const unsigned short* Arow1 = Wh_l + (wave * 16 + q) * LROW;
  const unsigned short* Arow2 = W2_l + (wave * 16 + q) * LROW;

  unsigned bar = 0;
  for (int t = 0; t < SEQ; ++t) {
    // previous output (bf16) -> LDS broadcast vector
    ((unsigned*)v_l)[tid] = ((const unsigned*)outb)[tid];
    __syncthreads();

    // -------- phase 1: h1 = tanh(pre_x[t] + Wh * out) --------
    v8f acc = {0.f,0.f,0.f,0.f,0.f,0.f,0.f,0.f};
#pragma unroll
    for (int kb = 0; kb < HID / 32; ++kb) {
      const int ka = kb * 32 + hi * 16;
      v16bf a = cat16(*(const v8us*)(Arow1 + ka), *(const v8us*)(Arow1 + ka + 8));
      const int kn = kb * 32 + hi * 16;
      v16bf b = cat16(*(const v8us*)(v_l + kn), *(const v8us*)(v_l + kn + 8));
      acc = __builtin_amdgcn_wmma_f32_16x16x32_bf16(false, a, false, b,
                                                    (short)0, acc, false, false);
    }
    {
      const int m = g * RPB + rowl;
      const float* pr = pre + (size_t)t * HID + m;
#pragma unroll
      for (int r = 0; r < 8; ++r) {
        float h = tanhf(pr[r] + acc[r]);
        if (q == 0) h1b[m + r] = f2us_bf(h);             // column-0 writers
      }
    }
    bar += NBLK; grid_barrier(counter, bar);

    ((unsigned*)v_l)[tid] = ((const unsigned*)h1b)[tid];
    __syncthreads();

    // Prefetch next step's pre_x row while phase 2 runs (L2-resident).
    {
      const int tn = (t + 1 < SEQ) ? (t + 1) : t;
      __builtin_prefetch(pre + (size_t)tn * HID + g * RPB + lane, 0, 0);
    }

    // -------- phase 2: out = tanh(W2 * h1 + c2) --------
    acc = (v8f){0.f,0.f,0.f,0.f,0.f,0.f,0.f,0.f};
#pragma unroll
    for (int kb = 0; kb < HID / 32; ++kb) {
      const int ka = kb * 32 + hi * 16;
      v16bf a = cat16(*(const v8us*)(Arow2 + ka), *(const v8us*)(Arow2 + ka + 8));
      const int kn = kb * 32 + hi * 16;
      v16bf b = cat16(*(const v8us*)(v_l + kn), *(const v8us*)(v_l + kn + 8));
      acc = __builtin_amdgcn_wmma_f32_16x16x32_bf16(false, a, false, b,
                                                    (short)0, acc, false, false);
    }
    {
#pragma unroll
      for (int r = 0; r < 8; ++r) {
        float h = tanhf(acc[r] + c2_l[rowl + r]);
        if (q == 0) outb[g * RPB + rowl + r] = f2us_bf(h);
      }
    }
    bar += NBLK; grid_barrier(counter, bar);
  }

  // -------- final fc: out[0] = dot(out, W_fc) + b_fc --------
  if (g == 0) {
    const int e = tid * 2;
    red[tid] = us2f_bf(outb[e]) * Wfc[e] + us2f_bf(outb[e + 1]) * Wfc[e + 1];
    __syncthreads();
    for (int s = 128; s > 0; s >>= 1) {
      if (tid < s) red[tid] += red[tid + s];
      __syncthreads();
    }
    if (tid == 0) out[0] = red[0] + bfc[0];
  }
}

// ---------------------------------------------------------------- host ----
extern "C" void kernel_launch(void* const* d_in, const int* in_sizes, int n_in,
                              void* d_out, int out_size, void* d_ws, size_t ws_size,
                              hipStream_t stream) {
  (void)in_sizes; (void)n_in; (void)out_size; (void)ws_size;
  const float* x     = (const float*)d_in[0];
  const float* W_ih1 = (const float*)d_in[1];
  const float* b_ih1 = (const float*)d_in[2];
  const float* b_hh1 = (const float*)d_in[3];
  const float* W_ih2 = (const float*)d_in[4];
  const float* b_ih2 = (const float*)d_in[5];
  const float* b_hh2 = (const float*)d_in[6];
  const float* W_fc  = (const float*)d_in[7];
  const float* b_fc  = (const float*)d_in[8];
  float* out = (float*)d_out;

  char* ws = (char*)d_ws;                       // needs ~50 MB
  unsigned*       counter = (unsigned*)(ws + 0);
  float*          c2      = (float*)(ws + 4096);
  unsigned short* outb    = (unsigned short*)(ws + 8192);
  unsigned short* h1b     = (unsigned short*)(ws + 12288);
  unsigned short* Wx      = (unsigned short*)(ws + 16384);
  unsigned short* Wh      = (unsigned short*)(ws + 16384 + 1 * 524288);
  unsigned short* W2      = (unsigned short*)(ws + 16384 + 2 * 524288);
  float*          pre     = (float*)(ws + 16384 + 3 * 524288);
  unsigned short* xbf     = (unsigned short*)(ws + 16384 + 3 * 524288
                                              + (size_t)SEQ * HID * 4);

  init_kernel<<<1, 512, 0, stream>>>(counter, (unsigned*)outb, c2, b_ih2, b_hh2);
  prep_w_kernel<<<(HID * HID) / 256, 256, 0, stream>>>(W_ih1, W_ih2, Wx, Wh, W2);
  prep_x_kernel<<<(SEQ * HID) / 256, 256, 0, stream>>>(x, xbf);
  gemm_prex<<<SEQ / 16, 256, 0, stream>>>(xbf, Wx, b_ih1, b_hh1, pre);

  static_assert(SMEM_SEQ <= 320 * 1024, "LDS budget");
  (void)hipFuncSetAttribute(reinterpret_cast<const void*>(rnn_seq),
                            hipFuncAttributeMaxDynamicSharedMemorySize, SMEM_SEQ);
  rnn_seq<<<NBLK, 256, SMEM_SEQ, stream>>>(pre, Wh, W2, c2, outb, h1b, counter,
                                           W_fc, b_fc, out);
}
